// MIXADModel_62070867361916
// MI455X (gfx1250) — matrix operand order repaired and checked by reading, not compiled, not run on hardware
//
#include <hip/hip_runtime.h>
#include <hip/hip_bf16.h>

// ---------------------------------------------------------------------------
// MIXAD graph-GRU seq2seq for MI455X (gfx1250, wave32, WMMA bf16)
// Persistent WGP-per-sequence design; all recurrent state in LDS; matmuls via
// v_wmma_f32_16x16x32_bf16 with f32 accumulation. Fragment loads are exactly
// two b128 loads each (ds_load_b128 / global_load_b128).
// ---------------------------------------------------------------------------

#define NNODE 256
#define BBATCH 32
#define TSTEP 128
#define HENC 64
#define MMEM 8
#define DMEM 32
#define HDEC 96
#define ENC_C 65      // IN_DIM + H
#define ENC_CPAD 96   // padded to multiple of 32
#define DEC_C 97      // OUT_DIM + DEC_H
#define DEC_CPAD 128

typedef __attribute__((ext_vector_type(16))) __bf16 v16bf;
typedef __attribute__((ext_vector_type(8)))  float  v8f;

// ---------------- bf16 <-> f32 helpers (ushort storage everywhere) ---------
static __device__ __forceinline__ unsigned short f2bf(float f) {
  unsigned int u = __float_as_uint(f);
  unsigned int r = (u + 0x7fffu + ((u >> 16) & 1u)) >> 16;
  return (unsigned short)r;
}
static __device__ __forceinline__ float bf2f(unsigned short h) {
  return __uint_as_float(((unsigned int)h) << 16);
}

// ---------------- hash-based Gumbel surrogate ------------------------------
static __device__ __forceinline__ unsigned int hash_u32(unsigned int a) {
  a ^= a >> 16; a *= 0x7feb352du;
  a ^= a >> 15; a *= 0x846ca68bu;
  a ^= a >> 16;
  return a;
}
static __device__ __forceinline__ float gumbelv(unsigned int key) {
  unsigned int h = hash_u32(key);
  float u = (float)(h) * 2.3283064e-10f;     // [0,1)
  u = fminf(fmaxf(u, 1e-7f), 1.0f - 1e-7f);
  return -__logf(-__logf(u));
}

// ---------------- WMMA fragment loaders (per CDNA5 ISA layouts) ------------
// A: 16x32 bf16, row-major source. Lane (row = lane%16, half = lane/16) needs
// K in [k0+8h, k0+8h+7] and [k0+8h+16, k0+8h+23]: two contiguous b128 loads.
static __device__ __forceinline__ v16bf load_A_frag(const unsigned short* A,
                                                    int lda, int row0, int k0) {
  union { v16bf v; uint4 q[2]; } f;
  int lane = threadIdx.x & 31;
  int half = lane >> 4;
  int row  = row0 + (lane & 15);
  const unsigned short* base = A + row * lda + k0 + 8 * half;
  f.q[0] = *(const uint4*)(base);
  f.q[1] = *(const uint4*)(base + 16);
  return f.v;
}
// B: 32x16 bf16 from TRANSPOSED storage BT[n][k] (K contiguous). Lane
// (col = lane%16, half = lane/16) needs K in [k0+8h, k0+8h+7] and
// [k0+8h+16, k0+8h+23]: two b128 loads.
static __device__ __forceinline__ v16bf load_B_fragT(const unsigned short* BT,
                                                     int ldb, int k0, int col0) {
  union { v16bf v; uint4 q[2]; } f;
  int lane = threadIdx.x & 31;
  int half = lane >> 4;
  int n    = col0 + (lane & 15);
  const unsigned short* base = BT + n * ldb + k0 + 8 * half;
  f.q[0] = *(const uint4*)(base);
  f.q[1] = *(const uint4*)(base + 16);
  return f.v;
}
// D tile (f32 acc) -> row-major bf16 store.
static __device__ __forceinline__ void store_D_bf16(unsigned short* D, int ldd,
                                                    int row0, int col0, v8f acc) {
  int lane  = threadIdx.x & 31;
  int rbase = row0 + 8 * (lane >> 4);
  int col   = col0 + (lane & 15);
#pragma unroll
  for (int r = 0; r < 8; ++r) D[(rbase + r) * ldd + col] = f2bf(acc[r]);
}

// ---------------------------------------------------------------------------
// One gconv+dense of a GRU cell:
//   out = ACT( sum_k  (S_k(256x256) @ cat(256xCPAD)) @ W_k(CPADxOUTW)  + b )
// catT: CPAD x 256 bf16 (transposed, K-contiguous), tmp: 256 x CPAD bf16.
// WT: OUTW x (6*CPAD) bf16 (transposed, padded). Dense accumulators live in
// VGPRs across the 6 Chebyshev terms; 8 waves own tile residue classes.
// ACT: 0 = sigmoid, 1 = tanh.
// ---------------------------------------------------------------------------
template <int CPAD, int OUTW, int ACT>
static __device__ void grucell_gemm(const unsigned short* __restrict__ Sall,
                                    const unsigned short* __restrict__ catT,
                                    unsigned short* __restrict__ tmp,
                                    const unsigned short* __restrict__ WT,
                                    const float* __restrict__ bias,
                                    unsigned short* __restrict__ outBF) {
  constexpr int NW  = 8;
  constexpr int CT  = OUTW / 16;      // dense col tiles
  constexpr int NT  = 16 * CT;        // dense total tiles (256 rows)
  constexpr int TPW = NT / NW;        // tiles per wave
  constexpr int DCT = CPAD / 16;      // diffusion col tiles
  const int wave = threadIdx.x >> 5;
  const int lane = threadIdx.x & 31;

  v8f acc[TPW];
#pragma unroll
  for (int i = 0; i < TPW; ++i) {
    int t = i * NW + wave;
    int ct = t % CT;
    float bv = bias[ct * 16 + (lane & 15)];
#pragma unroll
    for (int r = 0; r < 8; ++r) acc[i][r] = bv;
  }

  for (int k6 = 0; k6 < 6; ++k6) {
    __syncthreads();                       // tmp free from previous phase
    const unsigned short* Sk = Sall + k6 * NNODE * NNODE;
    __builtin_prefetch(Sall + ((k6 + 1) % 6) * NNODE * NNODE, 0, 0);
    // ---- diffusion: tmp = S_k @ cat -----------------------------------
    for (int t = wave; t < 16 * DCT; t += NW) {
      int rt = t / DCT, ct = t % DCT;
      v8f d;
#pragma unroll
      for (int r = 0; r < 8; ++r) d[r] = 0.0f;
      for (int kk = 0; kk < NNODE; kk += 32) {
        v16bf a = load_A_frag(Sk, NNODE, rt * 16, kk);
        v16bf b = load_B_fragT(catT, NNODE, kk, ct * 16);
        d = __builtin_amdgcn_wmma_f32_16x16x32_bf16(false, a, false, b,
                                                    (short)0, d, false, false);
      }
      store_D_bf16(tmp, CPAD, rt * 16, ct * 16, d);
    }
    __syncthreads();
    // ---- dense accumulate: acc += tmp @ W_k ---------------------------
#pragma unroll
    for (int i = 0; i < TPW; ++i) {
      int t = i * NW + wave;
      int rt = t / CT, ct = t % CT;
      for (int kk = 0; kk < CPAD; kk += 32) {
        v16bf a = load_A_frag(tmp, CPAD, rt * 16, kk);
        v16bf b = load_B_fragT(WT, 6 * CPAD, k6 * CPAD + kk, ct * 16);
        acc[i] = __builtin_amdgcn_wmma_f32_16x16x32_bf16(false, a, false, b,
                                                         (short)0, acc[i],
                                                         false, false);
      }
    }
  }
  __syncthreads();                         // all tmp reads done (out may alias)
#pragma unroll
  for (int i = 0; i < TPW; ++i) {
    int t = i * NW + wave;
    int rt = t / CT, ct = t % CT;
#pragma unroll
    for (int r = 0; r < 8; ++r) {
      float v = acc[i][r];
      v = (ACT == 0) ? (1.0f / (1.0f + __expf(-v))) : tanhf(v);
      int row = rt * 16 + 8 * (lane >> 4) + r;
      int col = ct * 16 + (lane & 15);
      outBF[row * OUTW + col] = f2bf(v);
    }
  }
}

// ---------------------------------------------------------------------------
// Prep kernels
// ---------------------------------------------------------------------------
__global__ void k_prep_embed(const float* We1, const float* We2,
                             const float* Mem, float* e1, float* e2) {
  int n = threadIdx.x;  // 256
  for (int d = 0; d < DMEM; ++d) {
    float a = 0.f, b = 0.f;
    for (int m = 0; m < MMEM; ++m) {
      a += We1[n * MMEM + m] * Mem[m * DMEM + d];
      b += We2[n * MMEM + m] * Mem[m * DMEM + d];
    }
    e1[n * DMEM + d] = a;
    e2[n * DMEM + d] = b;
  }
}

__global__ __launch_bounds__(256) void k_prep_lap(const float* e1,
                                                  const float* e2,
                                                  float* Lbuf) {
  __shared__ float nA[256], nB[256], red0[256], red1[256];
  int lap = blockIdx.x;                    // 0: (e1,e2), 1: (e2,e1)
  const float* A  = (lap == 0) ? e1 : e2;
  const float* Bm = (lap == 0) ? e2 : e1;
  int r = threadIdx.x;
  float sa = 0.f, sb = 0.f;
  for (int d = 0; d < DMEM; ++d) {
    float xa = A[r * DMEM + d];  sa += xa * xa;
    float xb = Bm[r * DMEM + d]; sb += xb * xb;
  }
  nA[r] = sqrtf(sa);
  nB[r] = sqrtf(sb);
  __syncthreads();
  float* L = Lbuf + lap * NNODE * NNODE;
  float deg = 0.f;
  for (int m = 0; m < NNODE; ++m) {
    float adj = 0.f;
    if (m != r) {
      float dot = 0.f;
      for (int d = 0; d < DMEM; ++d) dot += A[r * DMEM + d] * Bm[m * DMEM + d];
      float g = dot / (nA[r] * nB[m] + 1e-6f);
      g = (g + 1.0f) * 0.5f;
      unsigned int base = (unsigned int)(lap * 0x9E3779B9u) ^
                          (unsigned int)((r * NNODE + m) * 2);
      float g0 = g + gumbelv(base);
      float g1 = (1.0f - g) + gumbelv(base + 1u);
      adj = (g0 >= g1) ? 1.0f : 0.0f;      // straight-through fwd == y_hard
    }
    L[r * NNODE + m] = -adj;
    deg += adj;
  }
  L[r * NNODE + r] = deg;
  float mx = -1e30f, mn = 1e30f;
  for (int m = 0; m < NNODE; ++m) {
    float v = L[r * NNODE + m];
    mx = fmaxf(mx, v);
    mn = fminf(mn, v);
  }
  red0[r] = mx; red1[r] = mn;
  __syncthreads();
  for (int s = 128; s > 0; s >>= 1) {
    if (r < s) {
      red0[r] = fmaxf(red0[r], red0[r + s]);
      red1[r] = fminf(red1[r], red1[r + s]);
    }
    __syncthreads();
  }
  float inv = 2.0f / (red0[0] - red1[0]);
  for (int m = 0; m < NNODE; ++m)
    L[r * NNODE + m] = L[r * NNODE + m] * inv - ((m == r) ? 1.0f : 0.0f);
}

__global__ __launch_bounds__(256) void k_prep_cheb(const float* Lbuf,
                                                   unsigned short* Sbf) {
  int lap = blockIdx.x >> 8;
  int row = blockIdx.x & 255;
  int col = threadIdx.x;
  const float* L = Lbuf + lap * NNODE * NNODE;
  float acc = 0.f;
  for (int k = 0; k < NNODE; ++k) acc += L[row * NNODE + k] * L[k * NNODE + col];
  int base = lap * 3 * NNODE * NNODE + row * NNODE + col;
  float eye = (row == col) ? 1.0f : 0.0f;
  Sbf[base]                     = f2bf(eye);                  // T0 = I
  Sbf[base + NNODE * NNODE]     = f2bf(L[row * NNODE + col]); // T1 = Lhat
  Sbf[base + 2 * NNODE * NNODE] = f2bf(2.0f * acc - eye);     // T2 = 2L^2 - I
}

// W (6*C x O) row-major f32  ->  WT (O x 6*CPAD) bf16, zero-padded.
__global__ void k_prep_wt(const float* W, unsigned short* WT,
                          int C, int CPAD, int O) {
  int total = O * 6 * CPAD;
  for (int idx = blockIdx.x * blockDim.x + threadIdx.x; idx < total;
       idx += gridDim.x * blockDim.x) {
    int o = idx / (6 * CPAD);
    int rem = idx % (6 * CPAD);
    int kb = rem / CPAD, c = rem % CPAD;
    float v = (c < C) ? W[(kb * C + c) * O + o] : 0.0f;
    WT[idx] = f2bf(v);
  }
}

// ---------------------------------------------------------------------------
// Encoder: one workgroup per batch sequence, T=128 GRU steps, state in LDS.
// ---------------------------------------------------------------------------
__global__ __launch_bounds__(256) void k_encoder(
    const float* __restrict__ x, const unsigned short* __restrict__ S,
    const unsigned short* __restrict__ WTg, const float* __restrict__ gb,
    const unsigned short* __restrict__ WTu, const float* __restrict__ ub,
    float* __restrict__ h_out) {
  extern __shared__ char smem[];
  unsigned short* hbf  = (unsigned short*)smem;          // 256 x 64
  unsigned short* catT = hbf + NNODE * HENC;             // 96 x 256
  unsigned short* tmp  = catT + ENC_CPAD * NNODE;        // 256 x 96
  unsigned short* zr   = tmp + NNODE * ENC_CPAD;         // 256 x 128
  float* xt = (float*)(zr + NNODE * 2 * HENC);           // 256

  int b = blockIdx.x, tid = threadIdx.x;
  for (int i = tid; i < NNODE * HENC; i += 256) hbf[i] = 0;
  __syncthreads();

  for (int t = 0; t < TSTEP; ++t) {
    xt[tid] = x[((size_t)b * TSTEP + t) * NNODE + tid];
    __syncthreads();
    // catT = [x | h]^T padded
    for (int i = tid; i < ENC_CPAD * NNODE; i += 256) {
      int c = i >> 8, m = i & 255;
      float v = 0.f;
      if (c == 0) v = xt[m];
      else if (c < 1 + HENC) v = bf2f(hbf[m * HENC + (c - 1)]);
      catT[i] = f2bf(v);
    }
    __syncthreads();
    grucell_gemm<ENC_CPAD, 2 * HENC, 0>(S, catT, tmp, WTg, gb, zr);
    __syncthreads();
    // catT = [x | z*h]^T
    for (int i = tid; i < ENC_CPAD * NNODE; i += 256) {
      int c = i >> 8, m = i & 255;
      float v = 0.f;
      if (c == 0) v = xt[m];
      else if (c < 1 + HENC) {
        float z = bf2f(zr[m * 2 * HENC + (c - 1)]);
        v = z * bf2f(hbf[m * HENC + (c - 1)]);
      }
      catT[i] = f2bf(v);
    }
    __syncthreads();
    grucell_gemm<ENC_CPAD, HENC, 1>(S, catT, tmp, WTu, ub, tmp);  // hc -> tmp
    __syncthreads();
    for (int i = tid; i < NNODE * HENC; i += 256) {
      int n = i >> 6, j = i & 63;
      float r  = bf2f(zr[n * 2 * HENC + HENC + j]);
      float h  = bf2f(hbf[i]);
      float hc = bf2f(tmp[n * HENC + j]);
      hbf[i] = f2bf(r * h + (1.0f - r) * hc);
    }
    __syncthreads();
  }
  for (int i = tid; i < NNODE * HENC; i += 256)
    h_out[(size_t)b * NNODE * HENC + i] = bf2f(hbf[i]);
}

// ---------------------------------------------------------------------------
// Memory attention: h_de = [h, softmax(hWq M^T) M]   (tiny, VALU)
// ---------------------------------------------------------------------------
__global__ void k_attn(const float* __restrict__ h, const float* __restrict__ Wq,
                       const float* __restrict__ Mem,
                       unsigned short* __restrict__ hde) {
  int idx = blockIdx.x * blockDim.x + threadIdx.x;  // B*N
  if (idx >= BBATCH * NNODE) return;
  const float* hr = h + (size_t)idx * HENC;
  float q[DMEM];
  for (int d = 0; d < DMEM; ++d) {
    float a = 0.f;
    for (int j = 0; j < HENC; ++j) a += hr[j] * Wq[j * DMEM + d];
    q[d] = a;
  }
  float sc[MMEM], mx = -1e30f;
  for (int m = 0; m < MMEM; ++m) {
    float a = 0.f;
    for (int d = 0; d < DMEM; ++d) a += q[d] * Mem[m * DMEM + d];
    sc[m] = a;
    mx = fmaxf(mx, a);
  }
  float se = 0.f;
  for (int m = 0; m < MMEM; ++m) { sc[m] = __expf(sc[m] - mx); se += sc[m]; }
  float inv = 1.0f / se;
  unsigned short* o = hde + (size_t)idx * HDEC;
  for (int j = 0; j < HENC; ++j) o[j] = f2bf(hr[j]);
  for (int d = 0; d < DMEM; ++d) {
    float a = 0.f;
    for (int m = 0; m < MMEM; ++m) a += sc[m] * Mem[m * DMEM + d];
    o[HENC + d] = f2bf(a * inv);
  }
}

// ---------------------------------------------------------------------------
// Decoder: T=128 autoregressive GRU steps + projection; writes output.
// ---------------------------------------------------------------------------
__global__ __launch_bounds__(256) void k_decoder(
    const unsigned short* __restrict__ hde, const unsigned short* __restrict__ S,
    const unsigned short* __restrict__ WTg, const float* __restrict__ gb,
    const unsigned short* __restrict__ WTu, const float* __restrict__ ub,
    const float* __restrict__ projw, const float* __restrict__ projb,
    float* __restrict__ out) {
  extern __shared__ char smem[];
  unsigned short* hbf  = (unsigned short*)smem;          // 256 x 96
  unsigned short* catT = hbf + NNODE * HDEC;             // 128 x 256
  unsigned short* tmp  = catT + DEC_CPAD * NNODE;        // 256 x 128
  unsigned short* zr   = tmp + NNODE * DEC_CPAD;         // 256 x 192
  float* go = (float*)(zr + NNODE * 2 * HDEC);           // 256

  int b = blockIdx.x, tid = threadIdx.x;
  for (int i = tid; i < NNODE * HDEC; i += 256)
    hbf[i] = hde[(size_t)b * NNODE * HDEC + i];
  go[tid] = 0.0f;
  __syncthreads();

  for (int t = 0; t < TSTEP; ++t) {
    for (int i = tid; i < DEC_CPAD * NNODE; i += 256) {
      int c = i >> 8, m = i & 255;
      float v = 0.f;
      if (c == 0) v = go[m];
      else if (c < 1 + HDEC) v = bf2f(hbf[m * HDEC + (c - 1)]);
      catT[i] = f2bf(v);
    }
    __syncthreads();
    grucell_gemm<DEC_CPAD, 2 * HDEC, 0>(S, catT, tmp, WTg, gb, zr);
    __syncthreads();
    for (int i = tid; i < DEC_CPAD * NNODE; i += 256) {
      int c = i >> 8, m = i & 255;
      float v = 0.f;
      if (c == 0) v = go[m];
      else if (c < 1 + HDEC) {
        float z = bf2f(zr[m * 2 * HDEC + (c - 1)]);
        v = z * bf2f(hbf[m * HDEC + (c - 1)]);
      }
      catT[i] = f2bf(v);
    }
    __syncthreads();
    grucell_gemm<DEC_CPAD, HDEC, 1>(S, catT, tmp, WTu, ub, tmp);  // hc -> tmp
    __syncthreads();
    for (int i = tid; i < NNODE * HDEC; i += 256) {
      int n = i / HDEC, j = i % HDEC;
      float r  = bf2f(zr[n * 2 * HDEC + HDEC + j]);
      float h  = bf2f(hbf[i]);
      float hc = bf2f(tmp[n * HDEC + j]);
      hbf[i] = f2bf(r * h + (1.0f - r) * hc);
    }
    __syncthreads();
    // projection go[n] = h[n,:] . proj_w + proj_b  (one node per thread)
    {
      float a = 0.f;
      for (int j = 0; j < HDEC; ++j) a += bf2f(hbf[tid * HDEC + j]) * projw[j];
      a += projb[0];
      go[tid] = a;
      out[((size_t)b * TSTEP + t) * NNODE + tid] = a;
    }
    __syncthreads();
  }
}

// ---------------------------------------------------------------------------
extern "C" void kernel_launch(void* const* d_in, const int* in_sizes, int n_in,
                              void* d_out, int out_size, void* d_ws,
                              size_t ws_size, hipStream_t stream) {
  (void)in_sizes; (void)n_in; (void)out_size; (void)ws_size;
  const float* x   = (const float*)d_in[0];
  const float* Mem = (const float*)d_in[1];
  const float* Wq  = (const float*)d_in[2];
  const float* We1 = (const float*)d_in[3];
  const float* We2 = (const float*)d_in[4];
  const float* egw = (const float*)d_in[5];
  const float* egb = (const float*)d_in[6];
  const float* euw = (const float*)d_in[7];
  const float* eub = (const float*)d_in[8];
  const float* dgw = (const float*)d_in[9];
  const float* dgb = (const float*)d_in[10];
  const float* duw = (const float*)d_in[11];
  const float* dub = (const float*)d_in[12];
  const float* pjw = (const float*)d_in[13];
  const float* pjb = (const float*)d_in[14];

  char* w = (char*)d_ws;
  auto alloc = [&](size_t bytes) -> char* {
    char* p = w;
    w += (bytes + 255) & ~(size_t)255;
    return p;
  };
  float* e1            = (float*)alloc(NNODE * DMEM * 4);
  float* e2            = (float*)alloc(NNODE * DMEM * 4);
  float* Lbuf          = (float*)alloc(2 * NNODE * NNODE * 4);
  unsigned short* Sbf  = (unsigned short*)alloc(6 * NNODE * NNODE * 2);
  unsigned short* eWTg = (unsigned short*)alloc(128 * 6 * ENC_CPAD * 2);
  unsigned short* eWTu = (unsigned short*)alloc(64 * 6 * ENC_CPAD * 2);
  unsigned short* dWTg = (unsigned short*)alloc(192 * 6 * DEC_CPAD * 2);
  unsigned short* dWTu = (unsigned short*)alloc(96 * 6 * DEC_CPAD * 2);
  float* hEnc          = (float*)alloc((size_t)BBATCH * NNODE * HENC * 4);
  unsigned short* hDe  = (unsigned short*)alloc((size_t)BBATCH * NNODE * HDEC * 2);

  k_prep_embed<<<1, 256, 0, stream>>>(We1, We2, Mem, e1, e2);
  k_prep_lap<<<2, 256, 0, stream>>>(e1, e2, Lbuf);
  k_prep_cheb<<<512, 256, 0, stream>>>(Lbuf, Sbf);
  k_prep_wt<<<288, 256, 0, stream>>>(egw, eWTg, ENC_C, ENC_CPAD, 128);
  k_prep_wt<<<144, 256, 0, stream>>>(euw, eWTu, ENC_C, ENC_CPAD, 64);
  k_prep_wt<<<576, 256, 0, stream>>>(dgw, dWTg, DEC_C, DEC_CPAD, 192);
  k_prep_wt<<<288, 256, 0, stream>>>(duw, dWTu, DEC_C, DEC_CPAD, 96);

  size_t encLDS =
      (size_t)(NNODE * HENC + ENC_CPAD * NNODE + NNODE * ENC_CPAD +
               NNODE * 2 * HENC) * 2 + NNODE * 4;
  size_t decLDS =
      (size_t)(NNODE * HDEC + DEC_CPAD * NNODE + NNODE * DEC_CPAD +
               NNODE * 2 * HDEC) * 2 + NNODE * 4;
  hipFuncSetAttribute((const void*)k_encoder,
                      hipFuncAttributeMaxDynamicSharedMemorySize, (int)encLDS);
  hipFuncSetAttribute((const void*)k_decoder,
                      hipFuncAttributeMaxDynamicSharedMemorySize, (int)decLDS);

  k_encoder<<<BBATCH, 256, encLDS, stream>>>(x, Sbf, eWTg, egb, eWTu, eub, hEnc);
  k_attn<<<BBATCH, 256, 0, stream>>>(hEnc, Wq, Mem, hDe);
  k_decoder<<<BBATCH, 256, decLDS, stream>>>(hDe, Sbf, dWTg, dgb, dWTu, dub,
                                             pjw, pjb, (float*)d_out);
}
